// Transformer_90426241450800
// MI455X (gfx1250) — matrix operand order repaired and checked by reading, not compiled, not run on hardware
//
#include <hip/hip_runtime.h>
#include <math.h>

// ---------------------------------------------------------------------------
// Transformer encoder-decoder forward for MI455X (gfx1250, wave32, WMMA).
// B=4, S=2048, V=512, D=256, H=4 (HD=64), FF=1024, L=3.
// All GEMMs: bf16 x bf16 -> f32 via v_wmma_f32_16x16x32_bf16, with B always
// pre-transposed to [N,K] so LDS staging is pure 16-byte async copies
// (global_load_async_to_lds_b128, ASYNCcnt-fenced), double-buffered so the
// next K-slice's global latency hides behind the current slice's WMMAs.
// ---------------------------------------------------------------------------

#define TB 4
#define TS 2048
#define TV 512
#define TD 256
#define TH 4
#define THD 64
#define TFF 1024
#define TL 3
#define NTOK ((long)TB * TS)

typedef __attribute__((ext_vector_type(16))) __bf16 v16bf;
typedef __attribute__((ext_vector_type(8)))  float  v8f;

__device__ __forceinline__ __bf16 f2bf(float f) {
  unsigned u = __builtin_bit_cast(unsigned, f);
  u += 0x7FFFu + ((u >> 16) & 1u);                 // round-to-nearest-even
  unsigned short s = (unsigned short)(u >> 16);
  return __builtin_bit_cast(__bf16, s);
}

// ---------------------------------------------------------------------------
// Tiled WMMA GEMM: C[M,N] = A[M,K] * B^T + bias, optional GELU.
// A: bf16 row-major (lda). B: bf16 [N,K] row-major (ldb) -- pre-transposed.
// Outputs: optional f32 (Cf,ldc) and/or bf16 (Cb,ldcb).
// Tile 64x64x32, 128 threads (4 waves), each wave owns a 32x32 patch.
// Requires M%64==0, N%64==0, K%32==0 (true for every GEMM in this model).
// Double-buffered async global->LDS staging: ASYNCcnt completes in order, so
// after issuing the next tile's 4 ops, waiting ASYNCcnt<=4 guarantees the
// current tile has landed while the next is still in flight.
// ---------------------------------------------------------------------------
#define BM 64
#define BN 64
#define BKT 32
#define BKP 40   // padded LDS row stride (elems); 80B keeps 16B alignment

__global__ __launch_bounds__(128) void gemm_bf16_wmma(
    const __bf16* __restrict__ A, long lda,
    const __bf16* __restrict__ Bm, long ldb,
    const float* __restrict__ bias,
    float* __restrict__ Cf, long ldc,
    __bf16* __restrict__ Cb, long ldcb,
    int K, int do_gelu)
{
  __shared__ __bf16 sA[2][BM * BKP];
  __shared__ __bf16 sB[2][BN * BKP];

  const int m0 = blockIdx.x * BM;
  const int n0 = blockIdx.y * BN;
  const int tid  = threadIdx.x;
  const int lane = tid & 31;
  const int wave = tid >> 5;
  const int wm = (wave >> 1) * 32;   // wave tile origin in M
  const int wn = (wave & 1) * 32;    // wave tile origin in N

  // Per-thread staging slot: row 0..63, 16-elem half 0/16 (32B per thread/tile).
  const int sr = tid >> 1;
  const int scol = (tid & 1) << 4;
  const __bf16* gA = A  + (long)(m0 + sr) * lda + scol;
  const __bf16* gB = Bm + (long)(n0 + sr) * ldb + scol;
  // Low 32 bits of a flat shared pointer == LDS byte offset (aperture decode).
  unsigned la[2], lb[2];
  la[0] = (unsigned)(size_t)(&sA[0][sr * BKP + scol]);
  la[1] = (unsigned)(size_t)(&sA[1][sr * BKP + scol]);
  lb[0] = (unsigned)(size_t)(&sB[0][sr * BKP + scol]);
  lb[1] = (unsigned)(size_t)(&sB[1][sr * BKP + scol]);

  // Issue one K-slice (4 async B128 copies: 2 for A, 2 for B) into buffer buf.
  auto stage = [&](int buf, int k0) {
    const unsigned long long pa = (unsigned long long)(size_t)(gA + k0);
    const unsigned long long pb = (unsigned long long)(size_t)(gB + k0);
    asm volatile(
        "global_load_async_to_lds_b128 %0, %1, off\n\t"
        "global_load_async_to_lds_b128 %0, %1, off offset:16\n\t"
        "global_load_async_to_lds_b128 %2, %3, off\n\t"
        "global_load_async_to_lds_b128 %2, %3, off offset:16"
        :: "v"(la[buf]), "v"(pa), "v"(lb[buf]), "v"(pb)
        : "memory");
  };

  v8f acc[2][2] = {};

  stage(0, 0);                                   // prologue: first slice
  for (int k0 = 0; k0 < K; k0 += BKT) {
    const int cur = (k0 >> 5) & 1;
    const bool more = (k0 + BKT) < K;
    if (more) stage(cur ^ 1, k0 + BKT);          // prefetch next slice
    // Current slice's 4 ops are the oldest outstanding; async completes
    // in order, so <=4 (next slice still in flight) means cur has landed.
    if (more) asm volatile("s_wait_asynccnt 0x4" ::: "memory");
    else      asm volatile("s_wait_asynccnt 0x0" ::: "memory");
    __syncthreads();

    // ---- per-lane fragments (ISA 7.12.2 layouts) ----
    const int ar  = lane & 15;
    const int kb  = (lane >> 4) << 3;   // A: K chunks {kb..kb+7, 16+kb..}
    const int kb2 = (lane >> 4) << 4;   // unused; B reads the same kb chunks as A
    v16bf af[2], bfr[2];
#pragma unroll
    for (int i = 0; i < 2; ++i) {
      const __bf16* base = &sA[cur][(wm + i * 16 + ar) * BKP];
      uint4* pv = (uint4*)&af[i];
      pv[0] = *(const uint4*)(base + kb);
      pv[1] = *(const uint4*)(base + 16 + kb);
    }
#pragma unroll
    for (int j = 0; j < 2; ++j) {
      const __bf16* base = &sB[cur][(wn + j * 16 + ar) * BKP] + kb;
      uint4* pv = (uint4*)&bfr[j];
      pv[0] = *(const uint4*)(base);
      pv[1] = *(const uint4*)(base + 16);
    }
#pragma unroll
    for (int i = 0; i < 2; ++i)
#pragma unroll
      for (int j = 0; j < 2; ++j)
        acc[i][j] = __builtin_amdgcn_wmma_f32_16x16x32_bf16(
            false, af[i], false, bfr[j], (short)0, acc[i][j], false, false);
    __syncthreads();   // protect buffer reuse at distance 2
  }

  // ---- epilogue: bias, gelu, dual-precision store ----
  const int rb = (lane >> 4) << 3;   // D-matrix: M = vgpr + 8*(lane>=16)
  const int cl = lane & 15;
#pragma unroll
  for (int i = 0; i < 2; ++i) {
#pragma unroll
    for (int j = 0; j < 2; ++j) {
      const int col = n0 + wn + j * 16 + cl;
      const float bv = bias ? bias[col] : 0.f;
#pragma unroll
      for (int v = 0; v < 8; ++v) {
        const int row = m0 + wm + i * 16 + rb + v;
        float x = acc[i][j][v] + bv;
        if (do_gelu) x = 0.5f * x * (1.f + erff(x * 0.70710678118654752f));
        if (Cf) Cf[(long)row * ldc + col] = x;
        if (Cb) Cb[(long)row * ldcb + col] = f2bf(x);
      }
    }
  }
}

// ---------------------------------------------------------------------------
// Weight prep: f32 [R,C] row-major -> bf16 [C,R] (transpose), 32x32 LDS tile.
// ---------------------------------------------------------------------------
__global__ __launch_bounds__(256) void wtrans_kernel(
    const float* __restrict__ in, __bf16* __restrict__ out, int R, int C)
{
  __shared__ __bf16 t[32][33];
  const int r0 = blockIdx.x * 32;
  const int c0 = blockIdx.y * 32;
  const int r = threadIdx.x >> 5;
  const int c = threadIdx.x & 31;
#pragma unroll
  for (int i = 0; i < 4; ++i)
    t[r + 8 * i][c] = f2bf(in[(long)(r0 + r + 8 * i) * C + c0 + c]);
  __syncthreads();
#pragma unroll
  for (int i = 0; i < 4; ++i)
    out[(long)(c0 + r + 8 * i) * R + r0 + c] = t[c][r + 8 * i];
}

// ---------------------------------------------------------------------------
// V head-transpose: bf16 [NTOK, D] -> bf16 [B*H][HD][S] so P.V runs as the
// vectorized NT GEMM path. 32x32 LDS tile; head blocks (64) align to tiles.
// ---------------------------------------------------------------------------
__global__ __launch_bounds__(256) void vhead_transpose_kernel(
    const __bf16* __restrict__ in, __bf16* __restrict__ out)
{
  __shared__ __bf16 t[32][33];
  const long n0 = (long)blockIdx.x * 32;   // token base (b*S + s0)
  const int  d0 = blockIdx.y * 32;         // feature base
  const int r = threadIdx.x >> 5;
  const int c = threadIdx.x & 31;
#pragma unroll
  for (int i = 0; i < 4; ++i)
    t[r + 8 * i][c] = in[(n0 + r + 8 * i) * TD + d0 + c];
  __syncthreads();
  const long b = n0 / TS;
  const long s = n0 % TS;
#pragma unroll
  for (int i = 0; i < 4; ++i) {
    const int d  = d0 + r + 8 * i;
    const int h  = d >> 6;            // / HD
    const int dh = d & 63;            // % HD
    out[((b * TH + h) * THD + dh) * TS + s + c] = t[c][r + 8 * i];
  }
}

// ---------------------------------------------------------------------------
// Embedding gather: out[t,d] = emb[tok[t],d] + pos[t%S,d]; f32 + bf16 copies.
// ---------------------------------------------------------------------------
__global__ __launch_bounds__(256) void embed_kernel(
    const int* __restrict__ tok, const float* __restrict__ emb,
    const float* __restrict__ pos, float* __restrict__ of,
    __bf16* __restrict__ ob)
{
  const long idx = (long)blockIdx.x * 256 + threadIdx.x;   // over NTOK*D
  const long t = idx >> 8;                                  // D == 256
  const int  d = (int)(idx & 255);
  const int  s = (int)(t & (TS - 1));                       // S is pow2
  const float v = emb[(long)tok[t] * TD + d] + pos[(long)s * TD + d];
  of[idx] = v;
  ob[idx] = f2bf(v);
}

// ---------------------------------------------------------------------------
// y = LayerNorm(a + r) * g + b; one wave32 per token (D=256 -> 8/lane).
// ---------------------------------------------------------------------------
__global__ __launch_bounds__(256) void resid_ln_kernel(
    const float* __restrict__ a, const float* __restrict__ r,
    const float* __restrict__ g, const float* __restrict__ b,
    float* __restrict__ of, __bf16* __restrict__ ob)
{
  const int wave = threadIdx.x >> 5, lane = threadIdx.x & 31;
  const long t = (long)blockIdx.x * 8 + wave;
  const long base = t * TD;
  float x[8];
  float s = 0.f, ss = 0.f;
#pragma unroll
  for (int i = 0; i < 8; ++i) {
    const int d = lane * 8 + i;
    x[i] = a[base + d] + r[base + d];
    s += x[i];
    ss += x[i] * x[i];
  }
#pragma unroll
  for (int o = 16; o > 0; o >>= 1) {
    s  += __shfl_xor(s, o, 32);
    ss += __shfl_xor(ss, o, 32);
  }
  const float mu  = s * (1.f / TD);
  const float var = ss * (1.f / TD) - mu * mu;
  const float inv = rsqrtf(var + 1e-5f);
#pragma unroll
  for (int i = 0; i < 8; ++i) {
    const int d = lane * 8 + i;
    const float y = (x[i] - mu) * inv * g[d] + b[d];
    of[base + d] = y;
    ob[base + d] = f2bf(y);
  }
}

// ---------------------------------------------------------------------------
// Row softmax over Sk=2048 with scale + optional causal mask -> bf16 probs.
// ---------------------------------------------------------------------------
__global__ __launch_bounds__(256) void softmax_kernel(
    const float* __restrict__ scores, __bf16* __restrict__ probs,
    int causal, float scale)
{
  const int q = blockIdx.x;
  const long base = (long)q * TS;
  const int tid = threadIdx.x;
  const int lane = tid & 31, wave = tid >> 5;
  const int valid = causal ? (q + 1) : TS;
  __shared__ float red[8];

  float v[8];
  float m = -3.0e38f;
#pragma unroll
  for (int j = 0; j < 8; ++j) {
    const int col = j * 256 + tid;
    v[j] = (col < valid) ? scores[base + col] * scale : -3.0e38f;
    m = fmaxf(m, v[j]);
  }
#pragma unroll
  for (int o = 16; o > 0; o >>= 1) m = fmaxf(m, __shfl_xor(m, o, 32));
  if (lane == 0) red[wave] = m;
  __syncthreads();
  m = red[0];
#pragma unroll
  for (int w = 1; w < 8; ++w) m = fmaxf(m, red[w]);

  float s = 0.f;
#pragma unroll
  for (int j = 0; j < 8; ++j) {
    const int col = j * 256 + tid;
    v[j] = (col < valid) ? __expf(v[j] - m) : 0.f;
    s += v[j];
  }
#pragma unroll
  for (int o = 16; o > 0; o >>= 1) s += __shfl_xor(s, o, 32);
  __syncthreads();
  if (lane == 0) red[wave] = s;
  __syncthreads();
  s = 0.f;
#pragma unroll
  for (int w = 0; w < 8; ++w) s += red[w];
  const float inv = 1.f / s;
#pragma unroll
  for (int j = 0; j < 8; ++j) {
    const int col = j * 256 + tid;
    probs[base + col] = f2bf(v[j] * inv);
  }
}

// ---------------------------------------------------------------------------
// Host orchestration.
// ---------------------------------------------------------------------------
extern "C" void kernel_launch(void* const* d_in, const int* in_sizes, int n_in,
                              void* d_out, int out_size, void* d_ws, size_t ws_size,
                              hipStream_t stream) {
  (void)in_sizes; (void)n_in; (void)out_size; (void)ws_size;

  // ---- unpack inputs in setup_inputs() insertion order ----
  int ii = 0;
  const int* src = (const int*)d_in[ii++];
  const int* tgt = (const int*)d_in[ii++];
  auto nextf = [&]() { return (const float*)d_in[ii++]; };
  const float* src_emb = nextf();
  const float* tgt_emb = nextf();
  const float* src_pos = nextf();
  const float* tgt_pos = nextf();

  struct Lin   { const float* w; const float* b; };
  struct LNp   { const float* g; const float* b; };
  struct AttnP { Lin q, k, v, o; };
  struct BlockP { AttnP sa; LNp ln1; Lin ff1, ff2; LNp ln3; AttnP ca; LNp ln2; };

  auto rdLin  = [&]() { Lin l; l.w = nextf(); l.b = nextf(); return l; };
  auto rdLN   = [&]() { LNp l; l.g = nextf(); l.b = nextf(); return l; };
  auto rdAttn = [&]() { AttnP a; a.q = rdLin(); a.k = rdLin(); a.v = rdLin(); a.o = rdLin(); return a; };

  BlockP enc[TL], dec[TL];
  for (int l = 0; l < TL; ++l) {
    enc[l].sa = rdAttn(); enc[l].ln1 = rdLN();
    enc[l].ff1 = rdLin(); enc[l].ff2 = rdLin(); enc[l].ln3 = rdLN();
  }
  for (int l = 0; l < TL; ++l) {   // dict insertion order: sa,ln1,ff1,ff2,ln3,ca,ln2
    dec[l].sa = rdAttn(); dec[l].ln1 = rdLN();
    dec[l].ff1 = rdLin(); dec[l].ff2 = rdLin(); dec[l].ln3 = rdLN();
    dec[l].ca = rdAttn(); dec[l].ln2 = rdLN();
  }
  Lin fc = rdLin();

  // ---- workspace carve-up ----
  char* wp = (char*)d_ws;
  auto alloc = [&](size_t bytes) -> void* {
    void* r = (void*)wp;
    wp += (bytes + 255) & ~(size_t)255;
    return r;
  };
  float*  h_f  = (float*)alloc(NTOK * TD * 4);
  __bf16* h_b  = (__bf16*)alloc(NTOK * TD * 2);
  float*  y_f  = (float*)alloc(NTOK * TD * 4);
  __bf16* y_b  = (__bf16*)alloc(NTOK * TD * 2);
  float*  t_f  = (float*)alloc(NTOK * TD * 4);
  __bf16* qb   = (__bf16*)alloc(NTOK * TD * 2);
  __bf16* kbuf = (__bf16*)alloc(NTOK * TD * 2);
  __bf16* vbuf = (__bf16*)alloc(NTOK * TD * 2);
  __bf16* vT   = (__bf16*)alloc(NTOK * TD * 2);   // [B*H][HD][S]
  __bf16* abuf = (__bf16*)alloc(NTOK * TD * 2);
  __bf16* h1b  = (__bf16*)alloc(NTOK * TFF * 2);
  float*  sc   = (float*)alloc((long)TS * TS * 4);
  __bf16* pb   = (__bf16*)alloc((long)TS * TS * 2);

  // ---- weight prep: transpose + bf16 ([din,dout] f32 -> [dout,din] bf16) ----
  struct LinB  { const __bf16* w; const float* b; };   // w is [dout,din]
  struct AttnB { LinB q, k, v, o; };
  struct BlockB { AttnB sa, ca; LinB ff1, ff2; };
  auto cvtLin = [&](Lin l, long din, long dout) {
    __bf16* o = (__bf16*)alloc(din * dout * 2);
    dim3 g((unsigned)(din / 32), (unsigned)(dout / 32));
    wtrans_kernel<<<g, 256, 0, stream>>>(l.w, o, (int)din, (int)dout);
    LinB r; r.w = o; r.b = l.b; return r;
  };
  auto cvtAttn = [&](const AttnP& a) {
    AttnB r; r.q = cvtLin(a.q, TD, TD); r.k = cvtLin(a.k, TD, TD);
    r.v = cvtLin(a.v, TD, TD); r.o = cvtLin(a.o, TD, TD); return r;
  };
  BlockB encB[TL], decB[TL];
  for (int l = 0; l < TL; ++l) {
    encB[l].sa  = cvtAttn(enc[l].sa);
    encB[l].ff1 = cvtLin(enc[l].ff1, TD, TFF);
    encB[l].ff2 = cvtLin(enc[l].ff2, TFF, TD);
  }
  for (int l = 0; l < TL; ++l) {
    decB[l].sa  = cvtAttn(dec[l].sa);
    decB[l].ca  = cvtAttn(dec[l].ca);
    decB[l].ff1 = cvtLin(dec[l].ff1, TD, TFF);
    decB[l].ff2 = cvtLin(dec[l].ff2, TFF, TD);
  }
  LinB fcB = cvtLin(fc, TD, TV);

  // ---- launch helpers ----
  auto gemm = [&](const __bf16* A, long lda,
                  const __bf16* Bm, long ldb,
                  const float* bias,
                  float* Cf, long ldc,
                  __bf16* Cb, long ldcb,
                  long M, long N, long K, int gelu) {
    dim3 grid((unsigned)(M / BM), (unsigned)(N / BN));
    gemm_bf16_wmma<<<grid, 128, 0, stream>>>(A, lda, Bm, ldb, bias,
                                             Cf, ldc, Cb, ldcb, (int)K, gelu);
  };
  auto lnorm = [&](const float* a, float* xf, __bf16* xb, const LNp& p) {
    resid_ln_kernel<<<(unsigned)(NTOK / 8), 256, 0, stream>>>(a, xf, p.g, p.b, xf, xb);
  };
  auto mha = [&](const AttnB& aw, const __bf16* xq, const __bf16* xkv,
                 int causal, float* out_f) {
    gemm(xq,  TD, aw.q.w, TD, aw.q.b, nullptr, 0, qb,   TD, NTOK, TD, TD, 0);
    gemm(xkv, TD, aw.k.w, TD, aw.k.b, nullptr, 0, kbuf, TD, NTOK, TD, TD, 0);
    gemm(xkv, TD, aw.v.w, TD, aw.v.b, nullptr, 0, vbuf, TD, NTOK, TD, TD, 0);
    vhead_transpose_kernel<<<dim3((unsigned)(NTOK / 32), TD / 32), 256, 0, stream>>>(vbuf, vT);
    for (int b = 0; b < TB; ++b) {
      for (int h = 0; h < TH; ++h) {
        const long off  = (long)b * TS * TD + (long)h * THD;
        const long voff = ((long)b * TH + h) * THD * TS;
        // scores = Q_bh (S x 64) * K_bh^T  -> f32 [S,S]  (B = K rows, NT)
        gemm(qb + off, TD, kbuf + off, TD, nullptr,
             sc, TS, nullptr, 0, TS, TS, THD, 0);
        softmax_kernel<<<TS, 256, 0, stream>>>(sc, pb, causal, 0.125f);
        // out_bh = P (S x S) * V_bh -> bf16 strided into [N,D]  (B = V^T rows)
        gemm(pb, TS, vT + voff, TS, nullptr,
             nullptr, 0, abuf + off, TD, TS, THD, TS, 0);
      }
    }
    gemm(abuf, TD, aw.o.w, TD, aw.o.b, out_f, TD, nullptr, 0, NTOK, TD, TD, 0);
  };
  auto ffn = [&](const BlockB& wb, const Lin& f1, const Lin& f2,
                 float* xf, __bf16* xb, const LNp& ln3) {
    gemm(xb,  TD,  wb.ff1.w, TD,  f1.b, nullptr, 0, h1b, TFF, NTOK, TFF, TD, 1);
    gemm(h1b, TFF, wb.ff2.w, TFF, f2.b, t_f, TD, nullptr, 0, NTOK, TD, TFF, 0);
    lnorm(t_f, xf, xb, ln3);
  };

  // ---- encoder ----
  embed_kernel<<<(unsigned)(NTOK * TD / 256), 256, 0, stream>>>(src, src_emb, src_pos, h_f, h_b);
  for (int l = 0; l < TL; ++l) {
    mha(encB[l].sa, h_b, h_b, /*causal=*/0, t_f);
    lnorm(t_f, h_f, h_b, enc[l].ln1);
    ffn(encB[l], enc[l].ff1, enc[l].ff2, h_f, h_b, enc[l].ln3);
  }

  // ---- decoder ----
  embed_kernel<<<(unsigned)(NTOK * TD / 256), 256, 0, stream>>>(tgt, tgt_emb, tgt_pos, y_f, y_b);
  for (int l = 0; l < TL; ++l) {
    mha(decB[l].sa, y_b, y_b, /*causal=*/1, t_f);
    lnorm(t_f, y_f, y_b, dec[l].ln1);
    mha(decB[l].ca, y_b, h_b, /*causal=*/0, t_f);
    lnorm(t_f, y_f, y_b, dec[l].ln2);
    ffn(decB[l], dec[l].ff1, dec[l].ff2, y_f, y_b, dec[l].ln3);
  }

  // ---- final projection to logits [B,S,V] f32 ----
  gemm(y_b, TD, fcB.w, TV, fc.b, (float*)d_out, TV, nullptr, 0, NTOK, TV, TD, 0);
}
